// SwinTransformerBlock_12128987644008
// MI455X (gfx1250) — matrix operand order repaired and checked by reading, not compiled, not run on hardware
//
#include <hip/hip_runtime.h>
#include <math.h>

typedef __bf16 bf16t;
typedef __attribute__((ext_vector_type(16))) __bf16 v16bf;
typedef __attribute__((ext_vector_type(8)))  __bf16 v8bf;
typedef __attribute__((ext_vector_type(8)))  float  v8f;

#define C_    256
#define NH_   8
#define SS_   4
#define SCALE_ 0.17677669529663687f

// ---------- helpers ----------
static __device__ __forceinline__ unsigned short f2bf(float f) {
  unsigned int u = __float_as_uint(f);
  u += 0x7FFFu + ((u >> 16) & 1u);
  return (unsigned short)(u >> 16);
}

static __device__ __forceinline__ v8f wmma_bf16(v16bf a, v16bf b, v8f c) {
  return __builtin_amdgcn_wmma_f32_16x16x32_bf16(false, a, false, b, (short)0, c,
                                                 false, false);
}

// A fragment: row-major [16 x >=32] bf16, leading dim lda (two 16B loads, free concat)
static __device__ __forceinline__ v16bf load_a(const bf16t* A, int lda) {
  int lane = threadIdx.x & 31;
  const bf16t* p = A + (lane & 15) * lda + ((lane >> 4) << 4);
  v8bf lo = *(const v8bf*)p;
  v8bf hi = *(const v8bf*)(p + 8);
  return __builtin_shufflevector(lo, hi, 0,1,2,3,4,5,6,7,8,9,10,11,12,13,14,15);
}

// B fragment from pre-packed tile (512 bf16 per 32x16 K-tile, lane-major)
static __device__ __forceinline__ v16bf load_bp(const bf16t* tile) {
  int lane = threadIdx.x & 31;
  return *(const v16bf*)(tile + (lane << 4));
}

// B fragment where logical B[k][n] = M[n][k]; M row-major [16 x 32] (contiguous 32B)
static __device__ __forceinline__ v16bf load_bT(const bf16t* M, int ldm) {
  int lane = threadIdx.x & 31;
  return *(const v16bf*)(M + (lane & 15) * ldm + ((lane >> 4) << 4));
}

// window-order row m -> token index (same map for partition and reverse)
static __device__ __forceinline__ int winrow_token(int m) {
  int win = m >> 6, n = m & 63;
  int b = win >> 6, wloc = win & 63;
  int hs = ((wloc >> 3) << 3) + (n >> 3);
  int ws = ((wloc & 7) << 3) + (n & 7);
  int h = (hs + SS_) & 63, w = (ws + SS_) & 63;
  return (b << 12) + (h << 6) + w;
}

static __device__ __forceinline__ int region(int x) {
  return (x < 56) ? 0 : (x < 60 ? 1 : 2);
}

// ---------- kernels ----------
// convert W(KxN) f32 -> bf16 packed into WMMA-B fragment tiles:
// tile t = (k/32)*(N/16) + (n/16); within tile: lane*16+i where
// lane = 16*(kin>=16) + (n&15), i = kin&15  (kin = k&31)
__global__ void k_pack_w(const float* __restrict__ src,
                         unsigned short* __restrict__ dst, int K, int Nn) {
  int idx = blockIdx.x * 256 + threadIdx.x;
  if (idx >= K * Nn) return;
  int tile = idx >> 9, within = idx & 511;
  int lane = within >> 4, i = within & 15;
  int ntn = Nn >> 4;
  int kb = tile / ntn, nt = tile - kb * ntn;
  int k = (kb << 5) + ((lane >> 4) << 4) + i;
  int col = (nt << 4) + (lane & 15);
  dst[idx] = f2bf(src[(long)k * Nn + col]);
}

// one wave per token; window_map!=0 -> roll+window gather from x
__global__ void k_layernorm(const float* __restrict__ x, const float* __restrict__ g,
                            const float* __restrict__ bta,
                            unsigned short* __restrict__ out, int window_map) {
  int wave = threadIdx.x >> 5, lane = threadIdx.x & 31;
  int m = blockIdx.x * 8 + wave;
  int tok = window_map ? winrow_token(m) : m;
  const float* p = x + (long)tok * C_;
  float v[8]; float s = 0.f, s2 = 0.f;
#pragma unroll
  for (int i = 0; i < 8; ++i) {
    float t = p[i * 32 + lane]; v[i] = t; s += t; s2 += t * t;
  }
#pragma unroll
  for (int o = 16; o >= 1; o >>= 1) { s += __shfl_xor(s, o, 32); s2 += __shfl_xor(s2, o, 32); }
  float mu = s * (1.0f / C_);
  float var = s2 * (1.0f / C_) - mu * mu;
  float rs = rsqrtf(var + 1e-5f);
  unsigned short* q = out + (long)m * C_;
#pragma unroll
  for (int i = 0; i < 8; ++i) {
    int c = i * 32 + lane;
    q[c] = f2bf((v[i] - mu) * rs * g[c] + bta[c]);
  }
}

// qkv GEMM: A(65536x256) x W(256x768,packed) + b -> q/k row-major, v packed (B-frag)
__global__ void k_gemm_qkv(const unsigned short* __restrict__ A,
                           const unsigned short* __restrict__ Wt,
                           const float* __restrict__ bias,
                           unsigned short* __restrict__ qo,
                           unsigned short* __restrict__ ko,
                           unsigned short* __restrict__ vo) {
  __shared__ unsigned short sA[64 * 256];
  int tid = threadIdx.x;
  int m0 = blockIdx.x << 6;
  int n0 = blockIdx.y << 8;          // 256 cols per block, grid.y = 3
  const uint4* src = (const uint4*)(A + (long)m0 * 256);
#pragma unroll 2
  for (int i = tid; i < 64 * 256 / 8; i += 256) ((uint4*)sA)[i] = src[i];
  __syncthreads();
  int wave = tid >> 5, lane = tid & 31;
  int rowb = (wave >> 2) << 5;               // 0 or 32
  int colb = n0 + ((wave & 3) << 6);         // 4 waves x 64 cols
  const v8f vzero = {0.f,0.f,0.f,0.f,0.f,0.f,0.f,0.f};
  v8f acc[2][4];
#pragma unroll
  for (int mi = 0; mi < 2; ++mi)
#pragma unroll
    for (int t = 0; t < 4; ++t) acc[mi][t] = vzero;
  for (int kb = 0; kb < 8; ++kb) {
    v16bf a0 = load_a((const bf16t*)&sA[(rowb) * 256 + kb * 32], 256);
    v16bf a1 = load_a((const bf16t*)&sA[(rowb + 16) * 256 + kb * 32], 256);
#pragma unroll
    for (int t = 0; t < 4; ++t) {
      v16bf b = load_bp((const bf16t*)Wt + ((long)(kb * 48 + (colb >> 4) + t) << 9));
      acc[0][t] = wmma_bf16(a0, b, acc[0][t]);
      acc[1][t] = wmma_bf16(a1, b, acc[1][t]);
    }
  }
  int cl = lane & 15, r0 = (lane >> 4) << 3;
#pragma unroll
  for (int mi = 0; mi < 2; ++mi) {
#pragma unroll
    for (int t = 0; t < 4; ++t) {
      int col = colb + t * 16 + cl;
      int mat = col >> 8, cc = col & 255;
      int head = cc >> 5, hd = cc & 31;
      float bv = bias[col];
#pragma unroll
      for (int r = 0; r < 8; ++r) {
        int m = m0 + rowb + mi * 16 + r0 + r;
        int win = m >> 6, n = m & 63;
        long base = (long)((win << 3) + head) * 2048;
        float val = acc[mi][t][r] + bv;
        if (mat == 0) {
          qo[base + n * 32 + hd] = f2bf(val);
        } else if (mat == 1) {
          ko[base + n * 32 + hd] = f2bf(val);
        } else {
          // pack V into PV B-fragment layout: tile=(n/32)*2+(hd/16)
          int kb2 = n >> 5, kw = n & 31, ct = hd >> 4, c0 = hd & 15;
          vo[base + ((kb2 << 1) + ct) * 512 + ((((kw >> 4) << 4) + c0) << 4) + (kw & 15)]
              = f2bf(val);
        }
      }
    }
  }
}

// one workgroup (4 waves) per (window, head)
__global__ void k_attn(const unsigned short* __restrict__ qb,
                       const unsigned short* __restrict__ kbuf,
                       const unsigned short* __restrict__ vb,
                       const float* __restrict__ rpb,
                       unsigned short* __restrict__ ao) {
  __shared__ unsigned short sP[64 * 64];
  int wh = blockIdx.x;
  int win = wh >> 3, head = wh & 7;
  int wloc = win & 63;
  int wr0 = (wloc >> 3) << 3;   // window origin (shifted grid)
  int wc0 = (wloc & 7) << 3;
  const bf16t* qp = (const bf16t*)qb + (long)wh * 2048;
  const bf16t* kp = (const bf16t*)kbuf + (long)wh * 2048;
  const bf16t* vp = (const bf16t*)vb + (long)wh * 2048;
  int wave = threadIdx.x >> 5, lane = threadIdx.x & 31;
  int qr0 = wave << 4;
  const v8f vzero = {0.f,0.f,0.f,0.f,0.f,0.f,0.f,0.f};

  // Q K^T  (K dim = HD = 32 -> 1 wmma per 16x16 tile)
  v16bf aq = load_a(qp + qr0 * 32, 32);
  v8f s[4];
#pragma unroll
  for (int t = 0; t < 4; ++t) s[t] = wmma_bf16(aq, load_bT(kp + t * 16 * 32, 32), vzero);

  int cl = lane & 15, r0h = (lane >> 4) << 3;
  // scale + relative-position bias + shift mask
#pragma unroll
  for (int t = 0; t < 4; ++t) {
    int mc = t * 16 + cl;           // key index
    int i2 = mc >> 3, j2 = mc & 7;
    int code2 = region(wr0 + i2) * 3 + region(wc0 + j2);
#pragma unroll
    for (int r = 0; r < 8; ++r) {
      int nr = qr0 + r0h + r;       // query index
      int i1 = nr >> 3, j1 = nr & 7;
      int idx = (i1 - i2 + 7) * 15 + (j1 - j2 + 7);
      float bias = rpb[idx * NH_ + head];
      int code1 = region(wr0 + i1) * 3 + region(wc0 + j1);
      float mask = (code1 == code2) ? 0.f : -100.f;
      s[t][r] = s[t][r] * SCALE_ + bias + mask;
    }
  }
  // softmax over the 64 keys (4 regs x 16 lanes within each half-wave)
#pragma unroll
  for (int r = 0; r < 8; ++r) {
    float mx = fmaxf(fmaxf(s[0][r], s[1][r]), fmaxf(s[2][r], s[3][r]));
#pragma unroll
    for (int o = 8; o >= 1; o >>= 1) mx = fmaxf(mx, __shfl_xor(mx, o, 32));
    float sum = 0.f;
#pragma unroll
    for (int t = 0; t < 4; ++t) { float e = __expf(s[t][r] - mx); s[t][r] = e; sum += e; }
#pragma unroll
    for (int o = 8; o >= 1; o >>= 1) sum += __shfl_xor(sum, o, 32);
    float inv = 1.0f / sum;
    int nr = qr0 + r0h + r;
#pragma unroll
    for (int t = 0; t < 4; ++t) sP[nr * 64 + t * 16 + cl] = f2bf(s[t][r] * inv);
  }
  __syncthreads();

  // P V  (64x64 @ 64x32), V pre-packed in B-fragment tiles
  v8f oacc[2] = {vzero, vzero};
#pragma unroll
  for (int kk = 0; kk < 2; ++kk) {
    v16bf ap = load_a((const bf16t*)&sP[qr0 * 64 + kk * 32], 64);
#pragma unroll
    for (int ct = 0; ct < 2; ++ct)
      oacc[ct] = wmma_bf16(ap, load_bp(vp + ((kk << 1) + ct) * 512), oacc[ct]);
  }
#pragma unroll
  for (int ct = 0; ct < 2; ++ct) {
    int col = head * 32 + ct * 16 + cl;
#pragma unroll
    for (int r = 0; r < 8; ++r) {
      int nr = qr0 + r0h + r;
      ao[(long)(win * 64 + nr) * 256 + col] = f2bf(oacc[ct][r]);
    }
  }
}

// proj GEMM fused with window-reverse + roll-back + residual (f32 x2 out)
__global__ void k_gemm_proj(const unsigned short* __restrict__ A,
                            const unsigned short* __restrict__ Wt,
                            const float* __restrict__ bias,
                            const float* __restrict__ xin,
                            float* __restrict__ x2) {
  __shared__ unsigned short sA[64 * 256];
  int tid = threadIdx.x;
  int m0 = blockIdx.x << 6;
  const uint4* src = (const uint4*)(A + (long)m0 * 256);
#pragma unroll 2
  for (int i = tid; i < 64 * 256 / 8; i += 256) ((uint4*)sA)[i] = src[i];
  __syncthreads();
  int wave = tid >> 5, lane = tid & 31;
  int rowb = (wave >> 2) << 5;
  int colb = (wave & 3) << 6;
  const v8f vzero = {0.f,0.f,0.f,0.f,0.f,0.f,0.f,0.f};
  v8f acc[2][4];
#pragma unroll
  for (int mi = 0; mi < 2; ++mi)
#pragma unroll
    for (int t = 0; t < 4; ++t) acc[mi][t] = vzero;
  for (int kb = 0; kb < 8; ++kb) {
    v16bf a0 = load_a((const bf16t*)&sA[(rowb) * 256 + kb * 32], 256);
    v16bf a1 = load_a((const bf16t*)&sA[(rowb + 16) * 256 + kb * 32], 256);
#pragma unroll
    for (int t = 0; t < 4; ++t) {
      v16bf b = load_bp((const bf16t*)Wt + ((long)(kb * 16 + (colb >> 4) + t) << 9));
      acc[0][t] = wmma_bf16(a0, b, acc[0][t]);
      acc[1][t] = wmma_bf16(a1, b, acc[1][t]);
    }
  }
  int cl = lane & 15, r0 = (lane >> 4) << 3;
#pragma unroll
  for (int mi = 0; mi < 2; ++mi) {
#pragma unroll
    for (int t = 0; t < 4; ++t) {
      int col = colb + t * 16 + cl;
      float bv = bias[col];
#pragma unroll
      for (int r = 0; r < 8; ++r) {
        int m = m0 + rowb + mi * 16 + r0 + r;
        long idx = (long)winrow_token(m) * 256 + col;
        x2[idx] = xin[idx] + acc[mi][t][r] + bv;
      }
    }
  }
}

// fused MLP: fc1 + exact GELU kept in LDS, then fc2 + residual -> out (f32)
__global__ void k_mlp(const unsigned short* __restrict__ h2,
                      const float* __restrict__ x2,
                      const unsigned short* __restrict__ w1,
                      const float* __restrict__ b1,
                      const unsigned short* __restrict__ w2,
                      const float* __restrict__ b2,
                      float* __restrict__ out) {
  __shared__ unsigned short sA[16 * 256];    // 8 KB
  __shared__ unsigned short sG[16 * 1024];   // 32 KB
  int tid = threadIdx.x;
  int m0 = blockIdx.x << 4;
  const uint4* src = (const uint4*)(h2 + (long)m0 * 256);
  for (int i = tid; i < 16 * 256 / 8; i += 256) ((uint4*)sA)[i] = src[i];
  __syncthreads();
  int wave = tid >> 5, lane = tid & 31;
  int cl = lane & 15, r0 = (lane >> 4) << 3;
  const v8f vzero = {0.f,0.f,0.f,0.f,0.f,0.f,0.f,0.f};
  // fc1 + gelu : 16 x 1024; wave covers 8 col tiles, A-frag reused 8x
  {
    v8f acc[8];
#pragma unroll
    for (int t = 0; t < 8; ++t) acc[t] = vzero;
    int nc0 = wave * 8;
    for (int kb = 0; kb < 8; ++kb) {
      v16bf a = load_a((const bf16t*)&sA[kb * 32], 256);
#pragma unroll
      for (int t = 0; t < 8; ++t) {
        v16bf b = load_bp((const bf16t*)w1 + ((long)(kb * 64 + nc0 + t) << 9));
        acc[t] = wmma_bf16(a, b, acc[t]);
      }
    }
#pragma unroll
    for (int t = 0; t < 8; ++t) {
      int col = (nc0 + t) * 16 + cl; float bv = b1[col];
#pragma unroll
      for (int r = 0; r < 8; ++r) {
        float xv = acc[t][r] + bv;
        xv = 0.5f * xv * (1.0f + erff(xv * 0.70710678118654752f));
        sG[(r0 + r) * 1024 + col] = f2bf(xv);
      }
    }
  }
  __syncthreads();
  // fc2 + residual : 16 x 256; wave covers 2 col tiles
  {
    v8f acc[2] = {vzero, vzero};
    int nc0 = wave * 2;
    for (int kb = 0; kb < 32; ++kb) {
      v16bf a = load_a((const bf16t*)&sG[kb * 32], 1024);
#pragma unroll
      for (int t = 0; t < 2; ++t) {
        v16bf b = load_bp((const bf16t*)w2 + ((long)(kb * 16 + nc0 + t) << 9));
        acc[t] = wmma_bf16(a, b, acc[t]);
      }
    }
#pragma unroll
    for (int t = 0; t < 2; ++t) {
      int col = (nc0 + t) * 16 + cl; float bv = b2[col];
#pragma unroll
      for (int r = 0; r < 8; ++r) {
        long idx = (long)(m0 + r0 + r) * 256 + col;
        out[idx] = x2[idx] + acc[t][r] + bv;
      }
    }
  }
}

// ---------- launch ----------
extern "C" void kernel_launch(void* const* d_in, const int* in_sizes, int n_in,
                              void* d_out, int out_size, void* d_ws, size_t ws_size,
                              hipStream_t stream) {
  const float* x      = (const float*)d_in[0];
  const float* qkv_w  = (const float*)d_in[1];
  const float* qkv_b  = (const float*)d_in[2];
  const float* proj_w = (const float*)d_in[3];
  const float* proj_b = (const float*)d_in[4];
  const float* rpb    = (const float*)d_in[5];
  const float* n1g    = (const float*)d_in[6];
  const float* n1b    = (const float*)d_in[7];
  const float* n2g    = (const float*)d_in[8];
  const float* n2b    = (const float*)d_in[9];
  const float* fc1_w  = (const float*)d_in[10];
  const float* fc1_b  = (const float*)d_in[11];
  const float* fc2_w  = (const float*)d_in[12];
  const float* fc2_b  = (const float*)d_in[13];
  float* out = (float*)d_out;

  char* ws = (char*)d_ws;
  const long SEG = 33554432;  // 32 MB activation segments
  unsigned short* wq = (unsigned short*)(ws);              // 256x768 bf16 packed
  unsigned short* wp = (unsigned short*)(ws + 393216);     // 256x256 bf16 packed
  unsigned short* w1 = (unsigned short*)(ws + 524288);     // 256x1024 bf16 packed
  unsigned short* w2 = (unsigned short*)(ws + 1048576);    // 1024x256 bf16 packed
  unsigned short* xw = (unsigned short*)(ws + 1572864);          // LN1 windows / reused as attn-out
  unsigned short* qb = (unsigned short*)(ws + 1572864 + SEG);    // q / reused as h2
  unsigned short* kb = (unsigned short*)(ws + 1572864 + 2*SEG);  // k / x2(f32) spans k+v
  unsigned short* vb = (unsigned short*)(ws + 1572864 + 3*SEG);  // v (packed B-frag tiles)
  unsigned short* ao = xw;
  unsigned short* h2 = qb;
  float* x2 = (float*)kb;

  k_pack_w<<<768, 256, 0, stream>>>(qkv_w, wq, 256, 768);
  k_pack_w<<<256, 256, 0, stream>>>(proj_w, wp, 256, 256);
  k_pack_w<<<1024, 256, 0, stream>>>(fc1_w, w1, 256, 1024);
  k_pack_w<<<1024, 256, 0, stream>>>(fc2_w, w2, 1024, 256);

  k_layernorm<<<8192, 256, 0, stream>>>(x, n1g, n1b, xw, 1);
  k_gemm_qkv<<<dim3(1024, 3), 256, 0, stream>>>(xw, wq, qkv_b, qb, kb, vb);
  k_attn<<<8192, 128, 0, stream>>>(qb, kb, vb, rpb, ao);
  k_gemm_proj<<<dim3(1024, 1), 256, 0, stream>>>(ao, wp, proj_b, x, x2);
  k_layernorm<<<8192, 256, 0, stream>>>(x2, n2g, n2b, h2, 0);
  k_mlp<<<4096, 256, 0, stream>>>(h2, x2, w1, fc1_b, w2, fc2_b, out);
}
